// OctaveGDN_54322746360135
// MI455X (gfx1250) — matrix operand order repaired and checked by reading, not compiled, not run on hardware
//
#include <hip/hip_runtime.h>
#include <hip/hip_bf16.h>

// ---------------- constants (mirror reference) ----------------
#define CH      192
#define C1      48
#define CL      (CH - C1)       // 144
#define NBATCH  8
#define NPIX    65536           // 256*256
#define NTILE   128             // pixels per block tile
#define TILES_PER_IMG (NPIX / NTILE)   // 512

#define REPARAM   (3.814697265625e-6f)          // 2^-18
#define PEDESTAL  (REPARAM * REPARAM)
#define GAMMA_BOUND REPARAM
// BETA_BOUND = sqrt(BETA_MIN + REPARAM^2 + PEDESTAL)
#define BETA_BOUND (1.00001455e-3f)

// padded LDS stride for tanh tile (64 banks x 4B): 196 % 16 == 4 -> conflict-free row reads
#define TSTRIDE 196

// fragment-ready gamma: [kk(6)][m(12)][lane(32)][16 halves] = 36864 halves = 73728 B
#define GFRAG_HALVES (6 * 12 * 32 * 16)
#define GFRAG_BYTES  (GFRAG_HALVES * 2)          // 73728
#define WS_COPY_BYTES (GFRAG_BYTES + CH * 4)     // + 2b vector = 74496 = 9312 * 8
#define WS_QWORDS     (WS_COPY_BYTES / 8)        // 9312

#define LDS_TANH_BYTES ((size_t)NTILE * TSTRIDE * 4)   // 100352
#define LDS_G_OFF      (LDS_TANH_BYTES)                // G lands here (32B aligned)
#define LDS_B2_OFF     (LDS_TANH_BYTES + GFRAG_BYTES)  // 174080
#define SMEM_BYTES     (LDS_TANH_BYTES + (size_t)WS_COPY_BYTES)  // 174848

typedef __attribute__((ext_vector_type(16))) _Float16 v16h;
typedef __attribute__((ext_vector_type(8)))  float    v8f;
typedef __attribute__((ext_vector_type(4)))  unsigned int v4u;
typedef __attribute__((ext_vector_type(8)))  int      v8i;
typedef __attribute__((ext_vector_type(4)))  int      v4i;

// ---------------- kernel 1: reparameterize + swizzle into A-fragment layout ----------------
// A-fragment 16x32 f16 layout written here (contiguous k): lane<16: M=lane, half h -> K=kbase+h;
// lane>=16: M=lane-16, K=kbase+16+h (same k order as the B fragment in octgdn_main).
__global__ __launch_bounds__(256) void octgdn_prep(const float* __restrict__ beta,
                                                   const float* __restrict__ gamma,
                                                   _Float16* __restrict__ gfrag_ws,
                                                   float* __restrict__ b2_ws)
{
    int i = blockIdx.x * 256 + threadIdx.x;
    if (i < GFRAG_HALVES) {
        int h    = i & 15;
        int lane = (i >> 4) & 31;
        int m    = (i >> 9) % 12;
        int kk   = i / (12 * 32 * 16);
        int o    = m * 16 + (lane & 15);
        int K    = kk * 32 + ((lane >> 4) << 4) + h;
        float v  = fmaxf(gamma[o * CH + K], GAMMA_BOUND);
        gfrag_ws[i] = (_Float16)(v * v - PEDESTAL);
    }
    if (i < CH) {
        float v = fmaxf(beta[i], BETA_BOUND);
        b2_ws[i] = 2.0f * (v * v - PEDESTAL);   // beta enters twice in reference
    }
}

// fast tanh: t = exp(-2|x|); tanh = sign(x) * (1-t) * rcp(1+t)
__device__ __forceinline__ float fast_tanhf(float x)
{
    float ax = fabsf(x);
    float e  = __expf(-2.0f * ax);
    float r  = (1.0f - e) * __builtin_amdgcn_rcpf(1.0f + e);
    return copysignf(r, x);
}

// ---------------- kernel 2: TDM-staged gamma, fused tanh -> WMMA GEMM -> rsq-normalize ----------------
__global__ __launch_bounds__(256, 1) void octgdn_main(const float* __restrict__ xh,
                                                      const float* __restrict__ xl,
                                                      const _Float16* __restrict__ gfrag_ws,
                                                      float* __restrict__ out)
{
    extern __shared__ char smem_raw[];
    float*    lds_tanh = (float*)smem_raw;                     // [NTILE][TSTRIDE] f32
    _Float16* lds_g    = (_Float16*)(smem_raw + LDS_G_OFF);    // fragment-ready f16
    float*    lds_b2   = (float*)(smem_raw + LDS_B2_OFF);      // [CH] f32

    const int t     = threadIdx.x;
    const int wave  = t >> 5;
    const int lane  = t & 31;
    const int batch = blockIdx.x >> 9;            // 512 tiles per image
    const int p0    = (blockIdx.x & 511) << 7;    // first pixel of tile

    // ---- wave 0: kick off TDM DMA of gamma fragments + 2b into LDS (flat 1D, 8B units) ----
    if (wave == 0) {
        unsigned long long ga = (unsigned long long)(uintptr_t)gfrag_ws;
        v4u g0;
        g0[0] = 1u;                                            // count=1 valid user D#
        g0[1] = (unsigned int)LDS_G_OFF;                       // lds_addr (dyn LDS starts at 0)
        g0[2] = (unsigned int)(ga & 0xFFFFFFFFu);              // global_addr[31:0]
        g0[3] = (unsigned int)((ga >> 32) & 0x01FFFFFFu)       // global_addr[56:32]
              | (2u << 30);                                    // type=2 ("image")
        v8i g1;
        g1[0] = (3 << 16);                                     // mask=0, data_size=8B
        g1[1] = (int)((WS_QWORDS & 0xFFFF) << 16);             // tensor_dim0[15:0] @ bits63:48
        g1[2] = (int)((WS_QWORDS >> 16) & 0xFFFF)              // tensor_dim0[31:16]
              | (1 << 16);                                     // tensor_dim1 = 1
        g1[3] = (int)((WS_QWORDS & 0xFFFF) << 16);             // tile_dim0 @ bits127:112
        g1[4] = 0;                                             // tile_dim1/2 unused
        g1[5] = (int)WS_QWORDS;                                // tensor_dim0_stride[31:0]
        g1[6] = 0;
        g1[7] = 0;
        v4i z4 = {0, 0, 0, 0};
        v8i z8 = {0, 0, 0, 0, 0, 0, 0, 0};
        __builtin_amdgcn_tensor_load_to_lds(g0, g1, z4, z4, z8, 0);
    }

    // ---- all waves: load x tile (coalesced), tanh -> LDS [pix][ch] (overlaps the DMA) ----
    for (int i = t; i < CH * NTILE; i += 256) {
        int c   = i >> 7;       // channel 0..191
        int pix = i & (NTILE - 1);
        float x;
        if (c < C1) x = xh[(((size_t)batch * C1 + c)        << 16) + p0 + pix];
        else        x = xl[(((size_t)batch * CL + (c - C1)) << 16) + p0 + pix];
        lds_tanh[pix * TSTRIDE + c] = fast_tanhf(x);
    }

    if (wave == 0) {
        __builtin_amdgcn_s_wait_tensorcnt(0);  // gamma DMA done
    }
    __syncthreads();

    // ---- GEMM: norm[192 x 128] = G[192 x 192] * tanh^2[192 x 128] via f16 WMMA ----
    const int n0   = wave << 4;          // this wave's 16-pixel column block
    const int lrow = lane & 15;
    const int lhi  = lane >> 4;          // 0 for lanes 0-15, 1 for lanes 16-31

    v8f acc[12] = {};                    // 12 M-tiles x 16x16 f32

    const int    pix  = n0 + lrow;       // B layout: N = lane%16
    const float* trow = &lds_tanh[pix * TSTRIDE];

    #pragma unroll
    for (int kk = 0; kk < 6; ++kk) {
        const int kbase = kk << 5;

        // B fragment (32x16 f16): lanes0-15 halves = K kbase..+15, lanes16-31 = kbase+16..+31
        v16h bfrag;
        const float* bp = trow + kbase + (lhi << 4);
        #pragma unroll
        for (int i = 0; i < 16; ++i) {
            float v = bp[i];
            bfrag[i] = (_Float16)(v * v);
        }

        // A fragments: one contiguous 32B chunk per lane (pre-swizzled by prep kernel)
        #pragma unroll
        for (int m = 0; m < 12; ++m) {
            const v16h* ga = (const v16h*)&lds_g[(((kk * 12 + m) << 5) + lane) << 4];
            v16h afrag = *ga;
            acc[m] = __builtin_amdgcn_wmma_f32_16x16x32_f16(
                         false, afrag, false, bfrag, (short)0, acc[m], false, false);
        }
    }

    // ---- epilogue: y = tanh * rsq(|norm + 2b|), scatter into concatenated (y_h, y_l) ----
    const size_t YH_TOTAL = (size_t)NBATCH * C1 * NPIX;
    #pragma unroll
    for (int m = 0; m < 12; ++m) {
        #pragma unroll
        for (int r = 0; r < 8; ++r) {
            int   o  = m * 16 + r + (lhi << 3);       // C/D layout: M = r (+8 hi half)
            float s  = acc[m][r] + lds_b2[o];
            float tv = lds_tanh[pix * TSTRIDE + o];
            float y  = tv * __builtin_amdgcn_rsqf(fabsf(s));
            size_t idx = (o < C1)
                ? ((((size_t)batch * C1 + o)        << 16) + p0 + pix)
                : (YH_TOTAL + (((size_t)batch * CL + (o - C1)) << 16) + p0 + pix);
            out[idx] = y;
        }
    }
}

// ---------------- launch ----------------
extern "C" void kernel_launch(void* const* d_in, const int* in_sizes, int n_in,
                              void* d_out, int out_size, void* d_ws, size_t ws_size,
                              hipStream_t stream)
{
    (void)in_sizes; (void)n_in; (void)out_size; (void)ws_size;

    const float* xh    = (const float*)d_in[0];
    const float* xl    = (const float*)d_in[1];
    const float* beta  = (const float*)d_in[2];
    const float* gamma = (const float*)d_in[3];
    float*       out   = (float*)d_out;

    _Float16* gfrag_ws = (_Float16*)d_ws;                        // 73728 B, fragment layout
    float*    b2_ws    = (float*)((char*)d_ws + GFRAG_BYTES);    // 192 f32 appended

    octgdn_prep<<<144, 256, 0, stream>>>(beta, gamma, gfrag_ws, b2_ws);

    (void)hipFuncSetAttribute((const void*)octgdn_main,
                              hipFuncAttributeMaxDynamicSharedMemorySize, (int)SMEM_BYTES);

    const int nblocks = NBATCH * TILES_PER_IMG;   // 4096 pixel tiles
    octgdn_main<<<nblocks, 256, SMEM_BYTES, stream>>>(xh, xl, gfrag_ws, out);
}
